// ChannelAttentionModule_18184891531383
// MI455X (gfx1250) — matrix operand order repaired and checked by reading, not compiled, not run on hardware
//
#include <hip/hip_runtime.h>

typedef __attribute__((ext_vector_type(16))) _Float16 v16h;
typedef __attribute__((ext_vector_type(8)))  _Float16 v8h;
typedef __attribute__((ext_vector_type(8)))  float    v8f;
typedef int nl_b128_t __attribute__((vector_size(16)));   // v4i32, 16 bytes

#define B_  4
#define C_  256
#define CI_ 32
#define N_  4096
#define L2E 1.44269504f

#if __has_builtin(__builtin_amdgcn_global_load_async_to_lds_b128) && \
    __has_builtin(__builtin_amdgcn_s_wait_asynccnt)
#define NL_ASYNC 1
#else
#define NL_ASYNC 0
#endif

#if NL_ASYNC
#define NL_ASYNC_CP16(gsrc, ldst)                                          \
  __builtin_amdgcn_global_load_async_to_lds_b128(                          \
      (__attribute__((address_space(1))) nl_b128_t*)(gsrc),                \
      (__attribute__((address_space(3))) nl_b128_t*)(ldst), 0, 0)
#endif

// ---- WMMA fragment loaders (CDNA5 16-bit layouts, ISA 7.12.2) ----
// A (16xK=32, f16): lane l = row l&15; lanes 0-15 hold K=0..7,16..23; lanes 16-31 hold K=8..15,24..31
__device__ __forceinline__ v16h load_fragA(const _Float16* rowptr, int lane) {
  const int koff = (lane < 16) ? 0 : 8;
  v8h lo = *(const v8h*)(rowptr + koff);
  v8h hi = *(const v8h*)(rowptr + 16 + koff);
  v16h r;
#pragma unroll
  for (int i = 0; i < 8; ++i) { r[i] = lo[i]; r[i + 8] = hi[i]; }
  return r;
}
// B (K=32x16, f16): lane l = column l&15; lanes 0-15 hold K=0..7,16..23; lanes 16-31 hold K=8..15,24..31
__device__ __forceinline__ v16h load_fragB(const _Float16* colptr, int lane) {
  const int koff = (lane < 16) ? 0 : 8;
  v8h lo = *(const v8h*)(colptr + koff);
  v8h hi = *(const v8h*)(colptr + 16 + koff);
  v16h r;
#pragma unroll
  for (int i = 0; i < 8; ++i) { r[i] = lo[i]; r[i + 8] = hi[i]; }
  return r;
}
// C/D tile: lane l = column l&15; VGPR v = row v (lanes<16) or v+8 (lanes>=16)
__device__ __forceinline__ void store_tile_f16(v8f acc, _Float16* dst) {
  v8h h;
#pragma unroll
  for (int v = 0; v < 8; ++v) h[v] = (_Float16)acc[v];
  *(v8h*)dst = h;
}

// ---- K0: precompute f16 W, BN scale/shift (w_b folded in) ----
__global__ void nl_prep(const float* w_w, const float* w_b, const float* gamma,
                        const float* beta, const float* mean, const float* var,
                        _Float16* wh, float* bnscale, float* bnshift) {
  int idx = blockIdx.x * 256 + threadIdx.x;
  if (idx < C_ * C_) wh[idx] = (_Float16)w_w[idx];
  if (idx < C_) {
    float s = gamma[idx] / sqrtf(var[idx] + 1e-5f);
    bnscale[idx] = s;
    bnshift[idx] = fmaf(w_b[idx] - mean[idx], s, beta[idx]);
  }
}

// ---- K1: theta/phi projections -> f16 [b][n][32] ----
__global__ void nl_qk(const float* x, const float* tw, const float* tb,
                      const float* pw, const float* pb,
                      _Float16* thetaT, _Float16* phiT) {
  int t = blockIdx.x * 256 + threadIdx.x;
  int i = t & (CI_ - 1);
  int n = (t >> 5) & (N_ - 1);
  int b = t >> 17;
  const float* xp  = x + (size_t)b * C_ * N_ + n;
  const float* twp = tw + i * C_;
  const float* pwp = pw + i * C_;
  float at = tb[i], ap = pb[i];
#pragma unroll 8
  for (int c = 0; c < C_; ++c) {
    float xv = xp[(size_t)c * N_];
    at = fmaf(twp[c], xv, at);
    ap = fmaf(pwp[c], xv, ap);
  }
  size_t o = ((size_t)b * N_ + n) * CI_ + i;
  thetaT[o] = (_Float16)at;
  phiT[o]   = (_Float16)ap;
}

// ---- K2: row-max of S = theta^T phi (WMMA K=32) ----
__global__ void nl_rowmax(const _Float16* thetaT, const _Float16* phiT, float* rowmax) {
  int lane = threadIdx.x;
  int nbase = blockIdx.x * 16;
  int b = blockIdx.y;
  v16h a = load_fragA(thetaT + ((size_t)b * N_ + nbase + (lane & 15)) * CI_, lane);
  float rm[8];
#pragma unroll
  for (int v = 0; v < 8; ++v) rm[v] = -3.0e38f;
  for (int mt = 0; mt < N_ / 16; ++mt) {
    v16h bf = load_fragB(phiT + ((size_t)b * N_ + mt * 16 + (lane & 15)) * CI_, lane);
    v8f s = {};
    s = __builtin_amdgcn_wmma_f32_16x16x32_f16(false, a, false, bf, (short)0, s, false, false);
#pragma unroll
    for (int v = 0; v < 8; ++v) rm[v] = fmaxf(rm[v], s[v]);
  }
#pragma unroll
  for (int off = 1; off < 16; off <<= 1) {
#pragma unroll
    for (int v = 0; v < 8; ++v) rm[v] = fmaxf(rm[v], __shfl_xor(rm[v], off, 32));
  }
  if ((lane & 15) == 0) {
    int r0 = nbase + ((lane >= 16) ? 8 : 0);
#pragma unroll
    for (int v = 0; v < 8; ++v) rowmax[(size_t)b * N_ + r0 + v] = rm[v];
  }
}

// ---- K3: P = exp(S - rowmax) f16 [b][m][n]; rowsum Z ----
__global__ void nl_pexp(const _Float16* thetaT, const _Float16* phiT,
                        const float* rowmax, float* rowsum, _Float16* P) {
  int lane = threadIdx.x;
  int nbase = blockIdx.x * 16;
  int b = blockIdx.y;
  v16h a = load_fragA(thetaT + ((size_t)b * N_ + nbase + (lane & 15)) * CI_, lane);
  int rrow = nbase + ((lane >= 16) ? 8 : 0);
  float rmx[8], zs[8];
#pragma unroll
  for (int v = 0; v < 8; ++v) {
    rmx[v] = rowmax[(size_t)b * N_ + rrow + v] * L2E;
    zs[v] = 0.f;
  }
  _Float16* Pb = P + (size_t)b * N_ * N_;
  for (int mt = 0; mt < N_ / 16; ++mt) {
    int m = mt * 16 + (lane & 15);
    v16h bf = load_fragB(phiT + ((size_t)b * N_ + m) * CI_, lane);
    v8f s = {};
    s = __builtin_amdgcn_wmma_f32_16x16x32_f16(false, a, false, bf, (short)0, s, false, false);
    v8h e16;
#pragma unroll
    for (int v = 0; v < 8; ++v) {
      float e = exp2f(fmaf(s[v], L2E, -rmx[v]));
      zs[v] += e;
      e16[v] = (_Float16)e;
    }
    *(v8h*)(Pb + (size_t)m * N_ + rrow) = e16;   // [m][n] layout, 16B aligned
  }
#pragma unroll
  for (int off = 1; off < 16; off <<= 1) {
#pragma unroll
    for (int v = 0; v < 8; ++v) zs[v] += __shfl_xor(zs[v], off, 32);
  }
  if ((lane & 15) == 0) {
#pragma unroll
    for (int v = 0; v < 8; ++v) rowsum[(size_t)b * N_ + rrow + v] = zs[v];
  }
}

// ---- K4: V' = x / Z  -> f16 [b][c][n] ----
__global__ void nl_vprime(const float* x, const float* rowsum, _Float16* Vp) {
  size_t idx = (size_t)blockIdx.x * 256 + threadIdx.x;
  int n = (int)(idx & (N_ - 1));
  int b = (int)(idx >> 20);           // C_*N_ = 2^20
  Vp[idx] = (_Float16)(x[idx] / rowsum[b * N_ + n]);
}

// ---- K5: mid = V' @ P  (34 GF f16 GEMM) ----
// 4-wave block, 64c x 64m tile, double-buffered async LDS staging (8KB/step),
// 16 wmma per K=32 step -> 32 FLOP/byte of global traffic.
__global__ __launch_bounds__(128) void nl_attn_gemm(const _Float16* __restrict__ Vp,
                                                    const _Float16* __restrict__ P,
                                                    _Float16* __restrict__ midT) {
  __shared__ _Float16 As[2][64 * 32];   // [buf][row c][K]
  __shared__ _Float16 Bs[2][64 * 32];   // [buf][row m][K]
  const int t = threadIdx.x;
  const int lane = t & 31;
  const int wave = t >> 5;
  const int m0 = blockIdx.x * 64;
  const int c0 = blockIdx.y * 64;
  const int b = blockIdx.z;
  const _Float16* va = Vp + (size_t)b * C_ * N_;
  const _Float16* pa = P + (size_t)b * N_ * N_;

  // staging: 128 threads x 32B -> one 64x32 f16 panel (2 threads per row)
  const int sr = t >> 1;             // staged row 0..63
  const int sc = (t & 1) * 16;       // f16 offset within row (32B chunk pair)
  const _Float16* ga = va + (size_t)(c0 + sr) * N_ + sc;
  const _Float16* gb = pa + (size_t)(m0 + sr) * N_ + sc;
  _Float16* la = &As[0][sr * 32 + sc];
  _Float16* lb = &Bs[0][sr * 32 + sc];

  auto issue = [&](int buf, int nk) {
    const _Float16* g0 = ga + nk;
    const _Float16* g1 = gb + nk;
    _Float16* l0 = la + buf * (64 * 32);
    _Float16* l1 = lb + buf * (64 * 32);
#if NL_ASYNC
    NL_ASYNC_CP16(g0,     l0);
    NL_ASYNC_CP16(g0 + 8, l0 + 8);
    NL_ASYNC_CP16(g1,     l1);
    NL_ASYNC_CP16(g1 + 8, l1 + 8);
#else
    *(v8h*)l0       = *(const v8h*)g0;
    *(v8h*)(l0 + 8) = *(const v8h*)(g0 + 8);
    *(v8h*)l1       = *(const v8h*)g1;
    *(v8h*)(l1 + 8) = *(const v8h*)(g1 + 8);
#endif
  };

  const int cl = ((wave & 1) * 32) + (lane & 15);   // local A row for this wave/lane
  const int ml = ((wave >> 1) * 32) + (lane & 15);  // local B row for this wave/lane
  v8f acc00 = {}, acc01 = {}, acc10 = {}, acc11 = {};

  issue(0, 0);
  for (int nk = 0; nk < N_; nk += 32) {
    const int cur = (nk >> 5) & 1;
#if NL_ASYNC
    __builtin_amdgcn_s_wait_asynccnt(0);
#endif
    __syncthreads();
    if (nk + 32 < N_) issue(cur ^ 1, nk + 32);
    const _Float16* Ap = &As[cur][0];
    const _Float16* Bp = &Bs[cur][0];
    v16h A0 = load_fragA(Ap + (size_t)cl * 32, lane);
    v16h A1 = load_fragA(Ap + (size_t)(cl + 16) * 32, lane);
    v16h B0 = load_fragB(Bp + (size_t)ml * 32, lane);
    v16h B1 = load_fragB(Bp + (size_t)(ml + 16) * 32, lane);
    acc00 = __builtin_amdgcn_wmma_f32_16x16x32_f16(false, A0, false, B0, (short)0, acc00, false, false);
    acc01 = __builtin_amdgcn_wmma_f32_16x16x32_f16(false, A0, false, B1, (short)0, acc01, false, false);
    acc10 = __builtin_amdgcn_wmma_f32_16x16x32_f16(false, A1, false, B0, (short)0, acc10, false, false);
    acc11 = __builtin_amdgcn_wmma_f32_16x16x32_f16(false, A1, false, B1, (short)0, acc11, false, false);
  }

  const int chalf = (lane >= 16) ? 8 : 0;
  const int mg = m0 + ml;                      // global m column (j = 0)
  const int cg = c0 + ((wave & 1) * 32);       // global c tile base
  _Float16* mb = midT + (size_t)b * N_ * C_;
  store_tile_f16(acc00, mb + (size_t)mg * C_ + cg + chalf);
  store_tile_f16(acc01, mb + (size_t)(mg + 16) * C_ + cg + chalf);
  store_tile_f16(acc10, mb + (size_t)mg * C_ + cg + 16 + chalf);
  store_tile_f16(acc11, mb + (size_t)(mg + 16) * C_ + cg + 16 + chalf);
}

// ---- K6: out = BN(W @ mid + w_b) + x  (2.1 GF GEMM + fused epilogue) ----
__global__ void nl_wconv_bn(const _Float16* wh, const _Float16* midT,
                            const float* bnscale, const float* bnshift,
                            const float* x, float* out) {
  int lane = threadIdx.x;
  int m0 = blockIdx.x * 16;
  int o0 = blockIdx.y * 16;
  int b = blockIdx.z;
  const _Float16* mb = midT + (size_t)b * N_ * C_ + (size_t)(m0 + (lane & 15)) * C_;
  const _Float16* ap = wh + (size_t)(o0 + (lane & 15)) * C_;
  v8f acc = {};
  for (int ck = 0; ck < C_; ck += 32) {
    v16h A = load_fragA(ap + ck, lane);
    v16h Bf = load_fragB(mb + ck, lane);
    acc = __builtin_amdgcn_wmma_f32_16x16x32_f16(false, A, false, Bf, (short)0, acc, false, false);
  }
  int obase = o0 + ((lane >= 16) ? 8 : 0);
  int mcol = m0 + (lane & 15);
#pragma unroll
  for (int v = 0; v < 8; ++v) {
    int o = obase + v;
    size_t idx = ((size_t)(b * C_ + o)) * N_ + mcol;
    out[idx] = fmaf(acc[v], bnscale[o], bnshift[o]) + x[idx];
  }
}

extern "C" void kernel_launch(void* const* d_in, const int* in_sizes, int n_in,
                              void* d_out, int out_size, void* d_ws, size_t ws_size,
                              hipStream_t stream) {
  const float* x       = (const float*)d_in[0];
  const float* theta_w = (const float*)d_in[1];
  const float* theta_b = (const float*)d_in[2];
  const float* phi_w   = (const float*)d_in[3];
  const float* phi_b   = (const float*)d_in[4];
  const float* w_w     = (const float*)d_in[5];
  const float* w_b     = (const float*)d_in[6];
  const float* gamma   = (const float*)d_in[7];
  const float* beta    = (const float*)d_in[8];
  const float* mean    = (const float*)d_in[9];
  const float* var     = (const float*)d_in[10];
  float* out = (float*)d_out;

  char* w = (char*)d_ws;
  size_t off = 0;
  _Float16* thetaT = (_Float16*)(w + off); off += (size_t)B_ * N_ * CI_ * 2;  // 1 MB
  _Float16* phiT   = (_Float16*)(w + off); off += (size_t)B_ * N_ * CI_ * 2;  // 1 MB
  _Float16* wh     = (_Float16*)(w + off); off += (size_t)C_ * C_ * 2;        // 128 KB
  float*    bnscale= (float*)   (w + off); off += (size_t)C_ * 4;
  float*    bnshift= (float*)   (w + off); off += (size_t)C_ * 4;
  float*    rowmax = (float*)   (w + off); off += (size_t)B_ * N_ * 4;        // 64 KB
  float*    rowsum = (float*)   (w + off); off += (size_t)B_ * N_ * 4;        // 64 KB
  _Float16* Vp     = (_Float16*)(w + off); off += (size_t)B_ * C_ * N_ * 2;   // 8 MB
  _Float16* midT   = (_Float16*)(w + off); off += (size_t)B_ * C_ * N_ * 2;   // 8 MB
  _Float16* P      = (_Float16*)(w + off); off += (size_t)B_ * N_ * N_ * 2;   // 128 MB (L2-resident)

  nl_prep<<<dim3(C_ * C_ / 256), dim3(256), 0, stream>>>(w_w, w_b, gamma, beta, mean, var,
                                                         wh, bnscale, bnshift);
  nl_qk<<<dim3(B_ * N_ * CI_ / 256), dim3(256), 0, stream>>>(x, theta_w, theta_b,
                                                             phi_w, phi_b, thetaT, phiT);
  nl_rowmax<<<dim3(N_ / 16, B_), dim3(32), 0, stream>>>(thetaT, phiT, rowmax);
  nl_pexp<<<dim3(N_ / 16, B_), dim3(32), 0, stream>>>(thetaT, phiT, rowmax, rowsum, P);
  nl_vprime<<<dim3(B_ * C_ * N_ / 256), dim3(256), 0, stream>>>(x, rowsum, Vp);
  nl_attn_gemm<<<dim3(N_ / 64, C_ / 64, B_), dim3(128), 0, stream>>>(Vp, P, midT);
  nl_wconv_bn<<<dim3(N_ / 16, C_ / 16, B_), dim3(32), 0, stream>>>(wh, midT, bnscale, bnshift, x, out);
}
